// GraphSAGELayer_83167746719882
// MI455X (gfx1250) — matrix operand order repaired and checked by reading, not compiled, not run on hardware
//
#include <hip/hip_runtime.h>

#define N_NODES 100000
#define N_EDGES 1600000
#define D 128
#define DPAD_H (D + 8)   // f16 LDS row pad: 272B stride -> 4-bank skew
#define DPAD_F (D + 4)   // f32 LDS row pad: 528B stride -> 4-bank skew

typedef __attribute__((ext_vector_type(16))) _Float16 v16h;
typedef __attribute__((ext_vector_type(8)))  float    v8f;

// ---------------------------------------------------------------------------
// Kernel 1: zero the scratch accumulators (neighbor sums + counts)
// ---------------------------------------------------------------------------
__global__ void sage_init_ws(float* __restrict__ ws, int n) {
  int i = blockIdx.x * blockDim.x + threadIdx.x;
  if (i < n) ws[i] = 0.0f;
}

// ---------------------------------------------------------------------------
// Kernel 1b: one-shot f32 -> f16 conversion of both weight matrices.
// 16384 elements each; runs once, removes all v_cvt from the GEMM hot loop.
// ---------------------------------------------------------------------------
__global__ void sage_cvt_w(const float* __restrict__ Ws, const float* __restrict__ Wa,
                           _Float16* __restrict__ Wsh, _Float16* __restrict__ Wah) {
  int i = blockIdx.x * blockDim.x + threadIdx.x;
  if (i < D * D) {
    Wsh[i] = (_Float16)Ws[i];
    Wah[i] = (_Float16)Wa[i];
  }
}

// ---------------------------------------------------------------------------
// Kernel 2: edge scatter.  One edge handled by 32 threads, 4 floats each.
// sums[row] += x[col]; cnts[row] += 1.  x (51MB) and sums (51MB) are
// L2-resident (192MB L2) so the no-return f32 atomics resolve in L2.
// ---------------------------------------------------------------------------
__global__ __launch_bounds__(256) void sage_scatter(
    const float* __restrict__ x, const int* __restrict__ ei,
    float* __restrict__ sums, float* __restrict__ cnts) {
  long long gid = (long long)blockIdx.x * blockDim.x + threadIdx.x;
  int e    = (int)(gid >> 5);
  int part = (int)(gid & 31);
  if (e >= N_EDGES) return;
  int r = ei[e];            // edge_index[0][e]  (destination / segment id)
  int c = ei[N_EDGES + e];  // edge_index[1][e]  (source node)
  const float4 v = *(const float4*)(x + (long long)c * D + part * 4);
  float* dst = sums + (long long)r * D + part * 4;
  atomicAdd(dst + 0, v.x);
  atomicAdd(dst + 1, v.y);
  atomicAdd(dst + 2, v.z);
  atomicAdd(dst + 3, v.w);
  if (part == 0) atomicAdd(cnts + r, 1.0f);
}

// ---------------------------------------------------------------------------
// Kernel 3: fused  mean -> dual GEMM (WMMA, shared f32 accumulator) -> bias ->
//           ReLU -> LayerNorm -> store.
// One block = 16 nodes, 256 threads = 8 waves; wave w owns columns [16w,16w+16).
// ---------------------------------------------------------------------------
__global__ __launch_bounds__(256) void sage_fused(
    const float* __restrict__ x, const float* __restrict__ sums,
    const float* __restrict__ cnts,
    const _Float16* __restrict__ Wah, const float* __restrict__ bagg,
    const _Float16* __restrict__ Wsh, const float* __restrict__ bself,
    const float* __restrict__ gamma, const float* __restrict__ beta,
    float* __restrict__ out) {
  __shared__ _Float16 sX[16][DPAD_H];   // x rows, f16
  __shared__ _Float16 sM[16][DPAD_H];   // neighbor-mean rows, f16
  __shared__ float    sH[16][DPAD_F];   // post-ReLU activations, f32
  __shared__ float    sPs[16][17];      // LN partial sums   (pad 17: no conflicts)
  __shared__ float    sPq[16][17];      // LN partial sum-of-squares
  __shared__ float    sMu[16], sRs[16];

  const int tid  = threadIdx.x;
  const int row0 = blockIdx.x * 16;

  // ---- stage x and mean tiles into LDS (f16) --------------------------------
  {
    int r    = tid >> 4;           // 0..15
    int c0   = (tid & 15) * 8;     // 0..120
    int node = row0 + r;
    float inv = 1.0f / (cnts[node] + 1e-8f);
    const float* xp = x    + (long long)node * D + c0;
    const float* sp = sums + (long long)node * D + c0;
#pragma unroll
    for (int j = 0; j < 8; ++j) {
      sX[r][c0 + j] = (_Float16)xp[j];
      sM[r][c0 + j] = (_Float16)(sp[j] * inv);
    }
  }
  __syncthreads();

  // ---- WMMA: acc = x @ Wself^T + mean @ Wagg^T ------------------------------
  const int lane = tid & 31;
  const int wave = tid >> 5;                 // 0..7 = output N-tile
  const int mA   = lane & 15;                // A: lane -> row M
  const int kbA  = (lane < 16) ? 0 : 16;     // A: K sub-base by lane half
  const int nloc = lane & 15;                // B/D: lane -> col N
  const int n    = wave * 16 + nloc;         // global output column
  const int kbB  = (lane < 16) ? 0 : 16;     // B: K sub-base by lane half

  v8f acc = {};
#pragma unroll
  for (int pass = 0; pass < 2; ++pass) {
    const _Float16 (*sA)[DPAD_H] = pass ? sM : sX;
    const _Float16* W = pass ? Wah : Wsh;
#pragma unroll
    for (int kt = 0; kt < 4; ++kt) {
      const int kb = kt * 32;
      v16h a, b;
#pragma unroll
      for (int i = 0; i < 4; ++i) {   // A 16x32 f16, contiguous k = 16h + s (matches B)
        a[2*i    ] = sA[mA][kb + kbA + 2*i];
        a[2*i + 1] = sA[mA][kb + kbA + 2*i + 1];
        a[2*i + 8] = sA[mA][kb + kbA + 8 + 2*i];
        a[2*i + 9] = sA[mA][kb + kbA + 8 + 2*i + 1];
      }
      // B[k][n] = W[n][k]; lane reads 16 contiguous f16 (32B -> 2x b128)
      const _Float16* wrow = W + n * D + kb + kbB;
#pragma unroll
      for (int j = 0; j < 16; ++j) b[j] = wrow[j];
      acc = __builtin_amdgcn_wmma_f32_16x16x32_f16(
          /*neg_a=*/false, a, /*neg_b=*/false, b,
          /*c_mod=*/(short)0, acc, /*reuse_a=*/false, /*reuse_b=*/false);
    }
  }

  // ---- bias + ReLU into LDS (C/D layout: M = v + 8*(lane>=16), N = lane&15) -
  {
    float bsum = bself[n] + bagg[n];
#pragma unroll
    for (int v = 0; v < 8; ++v) {
      int m = v + ((lane < 16) ? 0 : 8);
      float h = acc[v] + bsum;
      sH[m][n] = h > 0.0f ? h : 0.0f;
    }
  }
  __syncthreads();

  // ---- LayerNorm statistics: 256-thread partial reduce, then 16-thread fold -
  {
    int r   = tid & 15;            // row
    int seg = tid >> 4;            // 16 segments of 8 columns
    float s = 0.0f, q = 0.0f;
#pragma unroll
    for (int j = 0; j < 8; ++j) {
      float v = sH[r][seg * 8 + j];
      s += v; q += v * v;
    }
    sPs[r][seg] = s;
    sPq[r][seg] = q;
  }
  __syncthreads();
  if (tid < 16) {
    float s = 0.0f, q = 0.0f;
#pragma unroll
    for (int g = 0; g < 16; ++g) { s += sPs[tid][g]; q += sPq[tid][g]; }
    float mu  = s * (1.0f / D);
    float var = q * (1.0f / D) - mu * mu;
    sMu[tid] = mu;
    sRs[tid] = rsqrtf(var + 1e-5f);
  }
  __syncthreads();

  // ---- normalize, affine, store --------------------------------------------
  {
    int r    = tid >> 4;
    int c0   = (tid & 15) * 8;
    int node = row0 + r;
    float mu = sMu[r], rs = sRs[r];
    float* op = out + (long long)node * D + c0;
#pragma unroll
    for (int j = 0; j < 8; ++j) {
      int c = c0 + j;
      op[j] = (sH[r][c] - mu) * rs * gamma[c] + beta[c];
    }
  }
}

// ---------------------------------------------------------------------------
extern "C" void kernel_launch(void* const* d_in, const int* in_sizes, int n_in,
                              void* d_out, int out_size, void* d_ws, size_t ws_size,
                              hipStream_t stream) {
  const float* x     = (const float*)d_in[0];
  const int*   ei    = (const int*)  d_in[1];
  const float* Wagg  = (const float*)d_in[2];
  const float* bagg  = (const float*)d_in[3];
  const float* Wself = (const float*)d_in[4];
  const float* bself = (const float*)d_in[5];
  const float* gamma = (const float*)d_in[6];
  const float* beta  = (const float*)d_in[7];
  float* out = (float*)d_out;

  // workspace layout: sums [N,128] f32 | cnts [N] f32 | Wself f16 | Wagg f16
  float*    sums = (float*)d_ws;
  float*    cnts = sums + (size_t)N_NODES * D;
  _Float16* Wsh  = (_Float16*)(cnts + N_NODES);       // 16B-aligned (51.6e6 % 16 == 0)
  _Float16* Wah  = Wsh + (size_t)D * D;

  int nz = N_NODES * D + N_NODES;
  sage_init_ws<<<(nz + 255) / 256, 256, 0, stream>>>(sums, nz);
  sage_cvt_w<<<(D * D + 255) / 256, 256, 0, stream>>>(Wself, Wagg, Wsh, Wah);

  long long sthreads = (long long)N_EDGES * 32;
  sage_scatter<<<(int)((sthreads + 255) / 256), 256, 0, stream>>>(x, ei, sums, cnts);

  sage_fused<<<N_NODES / 16, 256, 0, stream>>>(x, sums, cnts, Wah, bagg,
                                               Wsh, bself, gamma, beta, out);
}
